// LaserLinear_59854664237878
// MI455X (gfx1250) — matrix operand order repaired and checked by reading, not compiled, not run on hardware
//
#include <hip/hip_runtime.h>
#include <cstdint>
#include <cstddef>

// ---------------------------------------------------------------------------
// LaserLinear on MI455X (gfx1250): out = input @ (U*sigma @ V^T)^T
// Low-rank path:  T = X @ (V*sigma)   then   out = T @ U^T
// bf16 WMMA (v_wmma_f32_16x16x32_bf16), f32 accumulate.
// ---------------------------------------------------------------------------

typedef __attribute__((ext_vector_type(16))) __bf16 v16bf;
typedef __attribute__((ext_vector_type(8)))  float  v8f;

#define MROWS 8192    // B*S = 4*2048
#define DIN   4096
#define DOUT  4096
#define QRANK 1024

// f32 -> bf16, round-to-nearest-even
__device__ __forceinline__ unsigned short f2bf(float f) {
    unsigned u = __builtin_bit_cast(unsigned, f);
    u += 0x7FFFu + ((u >> 16) & 1u);
    return (unsigned short)(u >> 16);
}

struct Q2 { uint4 a, b; };   // 32 bytes == one v16bf fragment

// ---------------------------------------------------------------------------
// Kernel: elementwise f32 -> bf16 pack (8 elements / thread, b128 in & out)
// ---------------------------------------------------------------------------
__global__ __launch_bounds__(256)
void pack_bf16_kernel(const float* __restrict__ in, unsigned short* __restrict__ out) {
    size_t i = ((size_t)blockIdx.x * 256 + threadIdx.x) * 8;
    float4 f0 = *reinterpret_cast<const float4*>(in + i);
    float4 f1 = *reinterpret_cast<const float4*>(in + i + 4);
    uint4 r;
    r.x = (unsigned)f2bf(f0.x) | ((unsigned)f2bf(f0.y) << 16);
    r.y = (unsigned)f2bf(f0.z) | ((unsigned)f2bf(f0.w) << 16);
    r.z = (unsigned)f2bf(f1.x) | ((unsigned)f2bf(f1.y) << 16);
    r.w = (unsigned)f2bf(f1.z) | ((unsigned)f2bf(f1.w) << 16);
    *reinterpret_cast<uint4*>(out + i) = r;
}

// ---------------------------------------------------------------------------
// Kernel: Bt1[n][k] = bf16( V[k][n] * sigma[n] )   (LDS-tiled transpose+scale)
// V: [DIN x QRANK] f32 row-major.  Bt1: [QRANK x DIN] bf16 row-major.
// block (32,8), grid (DIN/32, QRANK/32)
// ---------------------------------------------------------------------------
__global__ __launch_bounds__(256)
void vt_scale_kernel(const float* __restrict__ V, const float* __restrict__ sigma,
                     unsigned short* __restrict__ Bt) {
    __shared__ float tile[32][33];
    const int k0 = blockIdx.x * 32;
    const int n0 = blockIdx.y * 32;
    const int tx = threadIdx.x;   // 0..31
    const int ty = threadIdx.y;   // 0..7
#pragma unroll
    for (int i = 0; i < 4; ++i) {
        int k = k0 + ty + 8 * i;
        tile[ty + 8 * i][tx] = V[(size_t)k * QRANK + (n0 + tx)];   // coalesced read
    }
    __syncthreads();
#pragma unroll
    for (int i = 0; i < 4; ++i) {
        int n = n0 + ty + 8 * i;
        Bt[(size_t)n * DIN + (k0 + tx)] = f2bf(tile[tx][ty + 8 * i] * sigma[n]); // coalesced write
    }
}

// ---------------------------------------------------------------------------
// GEMM: C[M x N] = A[M x K] * Bt[N x K]^T      (bf16 in, f32 accum)
// Workgroup: 256 threads = 8 waves; WG tile 128(M) x 128(N).
// Wave tile 32(M) x 64(N) = 2x4 WMMA 16x16 tiles; K step 32.
// ---------------------------------------------------------------------------
template <int N_DIM, int K_DIM, bool OUT_BF16>
__global__ __launch_bounds__(256)
void gemm_bf16_wmma(const unsigned short* __restrict__ A,
                    const unsigned short* __restrict__ Bt,
                    void* __restrict__ Cout) {
    const int lane  = threadIdx.x & 31;        // wave32
    const int wave  = threadIdx.x >> 5;        // 0..7
    const int m0    = blockIdx.x * 128 + (wave & 3) * 32;
    const int n0    = blockIdx.y * 128 + (wave >> 2) * 64;
    const int laneM = lane & 15;
    const int hi    = lane >> 4;               // 0: lanes 0-15, 1: lanes 16-31

    v8f acc[2][4] = {};

#pragma unroll 2
    for (int k0 = 0; k0 < K_DIM; k0 += 32) {
        v16bf afrag[2], bfrag[4];

        // A fragment, ISA 16-bit A 16x32 layout:
        //   lanes 0-15 : M=lane,   K = {k0..k0+7, k0+16..k0+23}
        //   lanes 16-31: M=lane-16,K = {k0+8..k0+15, k0+24..k0+31}
#pragma unroll
        for (int i = 0; i < 2; ++i) {
            const unsigned short* p =
                A + (size_t)(m0 + i * 16 + laneM) * K_DIM + (k0 + hi * 8);
            Q2 t;
            t.a = *reinterpret_cast<const uint4*>(p);
            t.b = *reinterpret_cast<const uint4*>(p + 16);
            __builtin_prefetch(p + 32, 0, 1);       // next K-step
            afrag[i] = __builtin_bit_cast(v16bf, t);
        }

        // B fragment (Bt is N-major): lane holds one column n, K order matches A
        //   lanes 0-15 : n = n0+j*16+lane,    K = {k0..k0+7, k0+16..k0+23}
        //   lanes 16-31: n = n0+j*16+lane-16, K = {k0+8..k0+15, k0+24..k0+31}
#pragma unroll
        for (int j = 0; j < 4; ++j) {
            const unsigned short* p =
                Bt + (size_t)(n0 + j * 16 + laneM) * K_DIM + (k0 + hi * 8);
            Q2 t;
            t.a = *reinterpret_cast<const uint4*>(p);
            t.b = *reinterpret_cast<const uint4*>(p + 16);
            bfrag[j] = __builtin_bit_cast(v16bf, t);
        }

#pragma unroll
        for (int i = 0; i < 2; ++i)
#pragma unroll
            for (int j = 0; j < 4; ++j)
                acc[i][j] = __builtin_amdgcn_wmma_f32_16x16x32_bf16(
                    false, afrag[i], false, bfrag[j],
                    (short)0, acc[i][j], false, false);
    }

    // C store, ISA 32-bit C/D 16x16 layout: VGPR v -> row m0+v (+8 for hi lanes)
#pragma unroll
    for (int i = 0; i < 2; ++i) {
#pragma unroll
        for (int j = 0; j < 4; ++j) {
            const int col  = n0 + j * 16 + laneM;
            const int rowb = m0 + i * 16 + hi * 8;
#pragma unroll
            for (int v = 0; v < 8; ++v) {
                const size_t idx = (size_t)(rowb + v) * N_DIM + col;
                if (OUT_BF16)
                    reinterpret_cast<unsigned short*>(Cout)[idx] = f2bf(acc[i][j][v]);
                else
                    reinterpret_cast<float*>(Cout)[idx] = acc[i][j][v];
            }
        }
    }
}

// ---------------------------------------------------------------------------
// Host launcher
// inputs: 0=input f32 [4,2048,4096], 1=U f32 [4096,1024],
//         2=sigma f32 [1024],        3=V f32 [4096,1024]
// output: f32 [4,2048,4096]
// ---------------------------------------------------------------------------
extern "C" void kernel_launch(void* const* d_in, const int* in_sizes, int n_in,
                              void* d_out, int out_size, void* d_ws, size_t ws_size,
                              hipStream_t stream) {
    (void)in_sizes; (void)n_in; (void)out_size; (void)ws_size;

    const float* X     = (const float*)d_in[0];
    const float* U     = (const float*)d_in[1];
    const float* sigma = (const float*)d_in[2];
    const float* V     = (const float*)d_in[3];
    float*       out   = (float*)d_out;

    // workspace carve-up (bf16 elements): X16 | T16 | BT1 | U16  (~101 MB)
    unsigned short* X16 = (unsigned short*)d_ws;
    unsigned short* T16 = X16 + (size_t)MROWS * DIN;
    unsigned short* BT1 = T16 + (size_t)MROWS * QRANK;
    unsigned short* U16 = BT1 + (size_t)QRANK * DIN;

    // 1) pack X -> bf16   (33.5M elems, 8/thread)
    pack_bf16_kernel<<<(MROWS * (size_t)DIN) / (8 * 256), 256, 0, stream>>>(X, X16);

    // 2) BT1[n][k] = bf16(V[k][n] * sigma[n])  (transpose + scale)
    vt_scale_kernel<<<dim3(DIN / 32, QRANK / 32), dim3(32, 8), 0, stream>>>(V, sigma, BT1);

    // 3) pack U -> bf16 (already [DOUT x QRANK] = N-major for GEMM2)
    pack_bf16_kernel<<<((size_t)DOUT * QRANK) / (8 * 256), 256, 0, stream>>>(U, U16);

    // 4) T = X @ Vs          [8192 x 4096] x [4096 x 1024] -> bf16
    gemm_bf16_wmma<QRANK, DIN, true>
        <<<dim3(MROWS / 128, QRANK / 128), 256, 0, stream>>>(X16, BT1, T16);

    // 5) out = T @ U^T       [8192 x 1024] x [1024 x 4096] -> f32
    gemm_bf16_wmma<DOUT, QRANK, false>
        <<<dim3(MROWS / 128, DOUT / 128), 256, 0, stream>>>(T16, U16, out);
}
